// SpanPairVAHead_56873956933783
// MI455X (gfx1250) — matrix operand order repaired and checked by reading, not compiled, not run on hardware
//
#include <hip/hip_runtime.h>
#include <hip/hip_bf16.h>
#include <math.h>

// Problem constants (from reference)
#define BB   64
#define LL   512
#define HH   1024
#define QQ   64
#define MM   (BB * QQ)      // 4096 rows
#define K1   (3 * HH)       // 3072
#define N1   256            // VA_H
#define N2   128            // VA_H/2

typedef _Float16 v16h __attribute__((ext_vector_type(16)));
typedef float    v8f  __attribute__((ext_vector_type(8)));

union U32H2 { unsigned u; _Float16 h[2]; };

__device__ __forceinline__ float gelu_exact(float x) {
    return 0.5f * x * (1.0f + erff(x * 0.70710678118654752440f));
}

// ---------------------------------------------------------------------------
// Convert W1 (K1 x N1 fp32, row-major) -> W1t (N1 x K1 fp16)  [B-matrix for GEMM1]
// ---------------------------------------------------------------------------
__global__ __launch_bounds__(256) void cvt_w1_kernel(const float* __restrict__ W1,
                                                     _Float16* __restrict__ W1t) {
    int idx = blockIdx.x * 256 + threadIdx.x;          // total K1*N1 = 786432
    if (idx >= K1 * N1) return;
    int k = idx / N1;
    int n = idx - k * N1;
    W1t[(size_t)n * K1 + k] = (_Float16)W1[idx];
}

// Convert W2 (N1 x N2 fp32) -> W2t (N2 x N1 fp16)  [B-matrix for GEMM2]
__global__ __launch_bounds__(256) void cvt_w2_kernel(const float* __restrict__ W2,
                                                     _Float16* __restrict__ W2t) {
    int idx = blockIdx.x * 256 + threadIdx.x;          // total N1*N2 = 32768
    if (idx >= N1 * N2) return;
    int k = idx / N2;
    int n = idx - k * N2;
    W2t[(size_t)n * N1 + k] = (_Float16)W2[idx];
}

// ---------------------------------------------------------------------------
// Span pooling: one block per (b,q). Produces fp16 A-matrix row
// [h_asp(1024) | h_opi(1024) | h_asp*h_opi(1024)]
// ---------------------------------------------------------------------------
__global__ __launch_bounds__(256) void pool_kernel(const float* __restrict__ hs,
                                                   const long long* __restrict__ spans,
                                                   const int* __restrict__ qmask,
                                                   _Float16* __restrict__ Ah) {
    const int m = blockIdx.x;            // b*Q + q
    const int b = m >> 6;                // Q = 64
    const int t = threadIdx.x;           // channel group: channels [4t, 4t+4)

    const long long s0 = spans[(size_t)m * 4 + 0];
    const long long e0 = spans[(size_t)m * 4 + 1];
    const long long s1 = spans[(size_t)m * 4 + 2];
    const long long e1 = spans[(size_t)m * 4 + 3];
    const int qm = (qmask[m] != 0);

    int lo0, hi0, lo1, hi1;
    if (s0 >= 2) { lo0 = (int)s0; hi0 = (int)e0; } else { lo0 = 1; hi0 = 1; }
    if (s1 >= 2) { lo1 = (int)s1; hi1 = (int)e1; } else { lo1 = 1; hi1 = 1; }
    if (!qm) { lo0 = 0; hi0 = -1; lo1 = 0; hi1 = -1; }   // empty -> zeros

    const size_t base = (size_t)b * LL * HH + (size_t)(4 * t);

    float a0 = 0.f, a1 = 0.f, a2 = 0.f, a3 = 0.f;
    for (int l = lo0; l <= hi0; ++l) {
        float4 v = *(const float4*)(hs + base + (size_t)l * HH);
        a0 += v.x; a1 += v.y; a2 += v.z; a3 += v.w;
    }
    float o0 = 0.f, o1 = 0.f, o2 = 0.f, o3 = 0.f;
    for (int l = lo1; l <= hi1; ++l) {
        float4 v = *(const float4*)(hs + base + (size_t)l * HH);
        o0 += v.x; o1 += v.y; o2 += v.z; o3 += v.w;
    }

    int c0 = hi0 - lo0 + 1; if (c0 < 1) c0 = 1;
    int c1 = hi1 - lo1 + 1; if (c1 < 1) c1 = 1;
    const float i0 = 1.0f / (float)c0;
    const float i1 = 1.0f / (float)c1;
    a0 *= i0; a1 *= i0; a2 *= i0; a3 *= i0;
    o0 *= i1; o1 *= i1; o2 *= i1; o3 *= i1;

    _Float16* row = Ah + (size_t)m * K1;
    const int c = 4 * t;
    row[c + 0]            = (_Float16)a0;
    row[c + 1]            = (_Float16)a1;
    row[c + 2]            = (_Float16)a2;
    row[c + 3]            = (_Float16)a3;
    row[HH + c + 0]       = (_Float16)o0;
    row[HH + c + 1]       = (_Float16)o1;
    row[HH + c + 2]       = (_Float16)o2;
    row[HH + c + 3]       = (_Float16)o3;
    row[2 * HH + c + 0]   = (_Float16)(a0 * o0);
    row[2 * HH + c + 1]   = (_Float16)(a1 * o1);
    row[2 * HH + c + 2]   = (_Float16)(a2 * o2);
    row[2 * HH + c + 3]   = (_Float16)(a3 * o3);
}

// ---------------------------------------------------------------------------
// Fused MLP head. 16 waves/block; block handles 16 rows of M.
// GEMM1 (K=3072) -> GELU -> GEMM2 (K=256) -> GELU -> GEMM3 (N=2) -> sigmoid
// ---------------------------------------------------------------------------
__global__ __launch_bounds__(512) void head_kernel(const _Float16* __restrict__ Ah,
                                                   const _Float16* __restrict__ W1t,
                                                   const _Float16* __restrict__ W2t,
                                                   const float* __restrict__ b1,
                                                   const float* __restrict__ b2,
                                                   const float* __restrict__ W3,
                                                   const float* __restrict__ b3,
                                                   const int* __restrict__ qmask,
                                                   float* __restrict__ out) {
    __shared__ _Float16 x1[16 * N1];   // 8 KB
    __shared__ float    x2[16 * N2];   // 8 KB

    const int tid  = threadIdx.x;
    const int w    = tid >> 5;         // wave id 0..15 (wave32)
    const int lane = tid & 31;
    const int m0   = blockIdx.x * 16;
    const int ml   = lane & 15;        // row (A) / col (B) within tile
    const int kh   = lane >> 4;        // K-half selector

    // ---- GEMM1: wave w computes output tile rows [m0,m0+16) x cols [16w,16w+16)
    v8f c = {};
    const _Float16* Arow = Ah  + (size_t)(m0 + ml) * K1;
    const _Float16* Brow = W1t + (size_t)(16 * w + ml) * K1;
    for (int kb = 0; kb < K1; kb += 32) {
        v16h a, bfrag;
        #pragma unroll
        for (int r = 0; r < 8; ++r) {
            // A (16x32 f16): VGPR r holds K = 2r + 8*(r>=4) + 8*kh (pairs contiguous)
            const int ka = kb + 2 * r + 8 * (r >> 2) + 8 * kh;
            U32H2 ua; ua.u = *(const unsigned*)(Arow + ka);
            a[2 * r] = ua.h[0]; a[2 * r + 1] = ua.h[1];
            // B (32x16 f16): VGPR r holds K = 2r + 8*(r>=4) + 8*kh (pairs contiguous in W1t)
            const int kbB = kb + 2 * r + 8 * (r >> 2) + 8 * kh;
            U32H2 ub; ub.u = *(const unsigned*)(Brow + kbB);
            bfrag[2 * r] = ub.h[0]; bfrag[2 * r + 1] = ub.h[1];
        }
        c = __builtin_amdgcn_wmma_f32_16x16x32_f16(false, a, false, bfrag,
                                                   (short)0, c, false, false);
    }
    // Epilogue 1: bias + exact GELU -> LDS fp16
    {
        const float bias = b1[16 * w + ml];
        const int rbase = kh * 8;
        #pragma unroll
        for (int v = 0; v < 8; ++v) {
            const int row = v + rbase;                 // C layout: M = v + 8*(lane>=16)
            x1[row * N1 + 16 * w + ml] = (_Float16)gelu_exact(c[v] + bias);
        }
    }
    __syncthreads();

    // ---- GEMM2: waves 0..7 compute cols [16w,16w+16) of N2=128
    v8f c2 = {};
    if (w < 8) {
        const _Float16* B2row = W2t + (size_t)(16 * w + ml) * N1;
        const _Float16* A2row = x1 + ml * N1;
        for (int kb = 0; kb < N1; kb += 32) {
            v16h a, bfrag;
            #pragma unroll
            for (int r = 0; r < 8; ++r) {
                const int ka = kb + 2 * r + 8 * (r >> 2) + 8 * kh;
                U32H2 ua; ua.u = *(const unsigned*)(A2row + ka);   // LDS b32 load
                a[2 * r] = ua.h[0]; a[2 * r + 1] = ua.h[1];
                const int kbB = kb + 2 * r + 8 * (r >> 2) + 8 * kh;
                U32H2 ub; ub.u = *(const unsigned*)(B2row + kbB);
                bfrag[2 * r] = ub.h[0]; bfrag[2 * r + 1] = ub.h[1];
            }
            c2 = __builtin_amdgcn_wmma_f32_16x16x32_f16(false, a, false, bfrag,
                                                        (short)0, c2, false, false);
        }
        const float bias = b2[16 * w + ml];
        const int rbase = kh * 8;
        #pragma unroll
        for (int v = 0; v < 8; ++v) {
            const int row = v + rbase;
            x2[row * N2 + 16 * w + ml] = gelu_exact(c2[v] + bias);
        }
    }
    __syncthreads();

    // ---- GEMM3 + sigmoid epilogue: 32 threads, one (row, col) each
    if (tid < 32) {
        const int row = tid >> 1;
        const int col = tid & 1;
        float s = b3[col];
        #pragma unroll 4
        for (int j = 0; j < N2; ++j)
            s += x2[row * N2 + j] * W3[j * 2 + col];
        const float va = 8.0f / (1.0f + expf(-s)) + 1.0f;
        const int m = m0 + row;
        out[m * 2 + col] = (qmask[m] != 0) ? va : 0.0f;
    }
}

// ---------------------------------------------------------------------------
extern "C" void kernel_launch(void* const* d_in, const int* in_sizes, int n_in,
                              void* d_out, int out_size, void* d_ws, size_t ws_size,
                              hipStream_t stream) {
    const float*     hs    = (const float*)d_in[0];
    const long long* spans = (const long long*)d_in[1];
    const int*       qmask = (const int*)d_in[2];
    const float*     W1    = (const float*)d_in[3];
    const float*     b1    = (const float*)d_in[4];
    const float*     W2    = (const float*)d_in[5];
    const float*     b2    = (const float*)d_in[6];
    const float*     W3    = (const float*)d_in[7];
    const float*     b3    = (const float*)d_in[8];
    float*           out   = (float*)d_out;

    // Workspace layout
    const size_t A_BYTES   = (size_t)MM * K1 * sizeof(_Float16);   // 24 MB
    const size_t W1T_BYTES = (size_t)N1 * K1 * sizeof(_Float16);   // 1.5 MB
    char* ws = (char*)d_ws;
    _Float16* Ah  = (_Float16*)ws;
    _Float16* W1t = (_Float16*)(ws + A_BYTES);
    _Float16* W2t = (_Float16*)(ws + A_BYTES + W1T_BYTES);

    cvt_w1_kernel<<<(K1 * N1 + 255) / 256, 256, 0, stream>>>(W1, W1t);
    cvt_w2_kernel<<<(N1 * N2 + 255) / 256, 256, 0, stream>>>(W2, W2t);
    pool_kernel<<<MM, 256, 0, stream>>>(hs, spans, qmask, Ah);
    head_kernel<<<MM / 16, 512, 0, stream>>>(Ah, W1t, W2t, b1, b2, W3, b3, qmask, out);
}
